// MultiEpisodeFusion_68075231642149
// MI455X (gfx1250) — matrix operand order repaired and checked by reading, not compiled, not run on hardware
//
#include <hip/hip_runtime.h>
#include <hip/hip_bf16.h>
#include <math.h>

typedef __attribute__((ext_vector_type(16))) _Float16 v16h;
typedef __attribute__((ext_vector_type(8)))  float    v8f;

#define Hh 64
#define Ww 64
#define HW 4096
#define Bn 2
#define CDIM 96
#define LTOT 12288          // 3*H*W
#define FEAT 786432         // Bn*CDIM*HW
#define SCALE_Q 0.2886751345948129f  // 12^-0.5
#define GEMM_WAVES 4        // waves per block in the GEMM kernel

// ---------------------------------------------------------------------------
// 1x1-conv GEMM via WMMA. Cin fixed = 96 (3 unrolled k-steps of 32).
// One wave -> 32(o) x 64(s) tile: 2 A fragments x 4 B fragments = 8 WMMA per
// k-step (24 static v_wmma). X panel (32k x 16s f32) is staged to LDS with
// async global->LDS DMA (ASYNCcnt) and B fragments are fed from LDS.
// Y[b,o,s] = sum_k W[o,k]*X[b,k,s] + bias[o] (+ Res[b,o,s])
// ---------------------------------------------------------------------------
__global__ void __launch_bounds__(128)
k_conv1x1_wmma(const float* __restrict__ X, const float* __restrict__ W,
               const float* __restrict__ bias, const float* __restrict__ Res,
               float* __restrict__ Y, int Cout)
{
    __shared__ float tile[GEMM_WAVES * 32 * 64];   // 32 KB: 8 KB per wave
    const int lane = threadIdx.x & 31;
    const int wiv  = threadIdx.x >> 5;
    const int wid  = blockIdx.x * GEMM_WAVES + wiv;
    const int tilesO = Cout >> 5;                  // 32-wide o tiles
    const int tilesS = HW >> 6;                    // 64-wide s tiles
    const int total  = Bn * tilesO * tilesS;
    if (wid >= total) return;                 // wave-uniform: EXEC all-1 for WMMA

    int b  = wid / (tilesO * tilesS);
    int r  = wid - b * (tilesO * tilesS);
    int ot = r / tilesS;
    int st = r - ot * tilesS;
    const int o0 = ot << 5, s0 = st << 6;
    const int half = lane >> 4, mn = lane & 15;

    const float* Xb = X + (size_t)b * CDIM * HW;
    float* wt = &tile[wiv * 2048];            // this wave's 32x64 panel

    v8f acc[2][4] = {};

#pragma unroll
    for (int kt = 0; kt < 3; ++kt) {
        const int k0 = kt * 32;

        // Two A fragments: rows o0+mn and o0+16+mn (16-bit A 16x32 layout)
        v16h a[2];
#pragma unroll
        for (int t = 0; t < 2; ++t) {
            const float* wrow = W + (size_t)(o0 + 16 * t + mn) * CDIM + k0;
#pragma unroll
            for (int v = 0; v < 8; ++v) {
                int kA = (v < 4) ? (8 * half + 2 * v) : (16 + 8 * half + 2 * (v - 4));
                a[t][2 * v]     = (_Float16)wrow[kA];
                a[t][2 * v + 1] = (_Float16)wrow[kA + 1];
            }
        }

        // Stage X[k0..k0+31][s0..s0+15] (f32) into LDS via async DMA:
        // 32 rows x 16 floats = 128 b128 chunks, 4 per lane.
#pragma unroll
        for (int it = 0; it < 4; ++it) {
            int idx = lane + 32 * it;
            int rr  = idx >> 2;
            int cc  = (idx & 3) << 2;
            unsigned la = (unsigned)(uintptr_t)&wt[rr * 64 + cc];
            unsigned long long ga = (unsigned long long)(uintptr_t)
                (Xb + (size_t)(k0 + rr) * HW + s0 + cc);
            asm volatile("global_load_async_to_lds_b128 %0, %1, off"
                         :: "v"(la), "v"(ga) : "memory");
        }
        asm volatile("s_wait_asynccnt 0x0" ::: "memory");

#pragma unroll
        for (int nf = 0; nf < 4; ++nf) {
            const int n0 = nf << 4;
            v16h bf;                           // B: col n; K = (v<4 ? 8*half+2v : 16+8*half+2(v-4)) + t
#pragma unroll
            for (int v = 0; v < 8; ++v) {
                int kB = (v < 4) ? (8 * half + 2 * v) : (16 + 8 * half + 2 * (v - 4));
                bf[2 * v]     = (_Float16)wt[kB * 64 + n0 + mn];
                bf[2 * v + 1] = (_Float16)wt[(kB + 1) * 64 + n0 + mn];
            }
#pragma unroll
            for (int t = 0; t < 2; ++t)
                acc[t][nf] = __builtin_amdgcn_wmma_f32_16x16x32_f16(
                    false, a[t], false, bf, (short)0, acc[t][nf], false, false);
        }
    }

    float* Yb = Y + (size_t)b * Cout * HW;
    const float* Rb = Res ? Res + (size_t)b * Cout * HW : nullptr;
#pragma unroll
    for (int t = 0; t < 2; ++t)
#pragma unroll
        for (int nf = 0; nf < 4; ++nf) {
            int s = s0 + (nf << 4) + mn;
#pragma unroll
            for (int v = 0; v < 8; ++v) {
                int o = o0 + 16 * t + v + 8 * half;   // C/D: VGPR v -> M=v+8*half
                float val = acc[t][nf][v] + bias[o];
                if (Rb) val += Rb[(size_t)o * HW + s];
                Yb[(size_t)o * HW + s] = val;
            }
        }
}

// ---------------------------------------------------------------------------
// Elementwise add over feature tensors: y = a + b
// ---------------------------------------------------------------------------
__global__ void k_add(const float* __restrict__ a, const float* __restrict__ b,
                      float* __restrict__ y, int n)
{
    int i = blockIdx.x * blockDim.x + threadIdx.x;
    if (i < n) y[i] = a[i] + b[i];
}

// ---------------------------------------------------------------------------
// Depthwise KxK conv, 64x64 image, arbitrary batch strides
// ---------------------------------------------------------------------------
__global__ void k_dwconv(const float* __restrict__ x, const float* __restrict__ w,
                         const float* __restrict__ bias, float* __restrict__ y,
                         int ks, int pad, long bstr_in, long bstr_out)
{
    int idx = blockIdx.x * blockDim.x + threadIdx.x;
    if (idx >= Bn * CDIM * HW) return;
    int j = idx & 63, i = (idx >> 6) & 63;
    int c = (idx >> 12) % CDIM;
    int b = idx / (CDIM * HW);
    const float* xc = x + (size_t)b * bstr_in + (size_t)c * HW;
    const float* wc = w + (size_t)c * ks * ks;
    float acc = bias[c];
    for (int u = 0; u < ks; ++u) {
        int ii = i + u - pad;
        if (ii < 0 || ii >= Hh) continue;
        for (int v = 0; v < ks; ++v) {
            int jj = j + v - pad;
            if (jj < 0 || jj >= Ww) continue;
            acc += wc[u * ks + v] * xc[ii * Ww + jj];
        }
    }
    y[(size_t)b * bstr_out + (size_t)c * HW + i * Ww + j] = acc;
}

// ---------------------------------------------------------------------------
// Per-channel GroupNorm over HW (block per (b,c)), optional exact GELU
// ---------------------------------------------------------------------------
__global__ void __launch_bounds__(256)
k_gn(const float* __restrict__ x, const float* __restrict__ g,
     const float* __restrict__ bt, float* __restrict__ y, int act)
{
    __shared__ float ssum[256], ssq[256];
    const int bc = blockIdx.x;                   // b*CDIM + c
    const float* xc = x + (size_t)bc * HW;
    float s = 0.f, q = 0.f;
    for (int t = threadIdx.x; t < HW; t += 256) { float v = xc[t]; s += v; q += v * v; }
    ssum[threadIdx.x] = s; ssq[threadIdx.x] = q;
    __syncthreads();
    for (int off = 128; off > 0; off >>= 1) {
        if ((int)threadIdx.x < off) {
            ssum[threadIdx.x] += ssum[threadIdx.x + off];
            ssq[threadIdx.x]  += ssq[threadIdx.x + off];
        }
        __syncthreads();
    }
    float mu  = ssum[0] * (1.f / HW);
    float var = ssq[0] * (1.f / HW) - mu * mu;
    float inv = rsqrtf(var + 1e-5f);
    int c = bc % CDIM;
    float gg = g[c], bb = bt[c];
    float* yc = y + (size_t)bc * HW;
    for (int t = threadIdx.x; t < HW; t += 256) {
        float v = (xc[t] - mu) * inv * gg + bb;
        if (act) v = 0.5f * v * (1.f + erff(v * 0.70710678118654752f));
        yc[t] = v;
    }
}

// ---------------------------------------------------------------------------
// LayerNorm over 96 channels per pixel; strided in/out; optional add + SiLU
// ---------------------------------------------------------------------------
__global__ void k_ln2d(const float* __restrict__ x, long xB, long xC,
                       const float* __restrict__ addp,
                       const float* __restrict__ g, const float* __restrict__ bt,
                       float* __restrict__ y, long yB, long yC, int do_silu)
{
    int idx = blockIdx.x * blockDim.x + threadIdx.x;
    if (idx >= Bn * HW) return;
    int p = idx & (HW - 1);
    int b = idx / HW;
    const float* xb = x + (size_t)b * xB + p;
    float s = 0.f, q = 0.f;
    for (int c = 0; c < CDIM; ++c) { float v = xb[(size_t)c * xC]; s += v; q += v * v; }
    float mu  = s * (1.f / CDIM);
    float var = q * (1.f / CDIM) - mu * mu;
    float inv = rsqrtf(var + 1e-6f);
    for (int c = 0; c < CDIM; ++c) {
        float v = (xb[(size_t)c * xC] - mu) * inv * g[c] + bt[c];
        if (addp) v += addp[(size_t)b * CDIM * HW + (size_t)c * HW + p];
        if (do_silu) v = v / (1.f + expf(-v));
        y[(size_t)b * yB + (size_t)c * yC + p] = v;
    }
}

// cross-scan position: scan step l of direction k -> (d,i,j) in (3,H,W)
__device__ __forceinline__ int scan_pos(int k, int l)
{
    int lp = (k < 3) ? l : (LTOT - 1 - l);
    int km = k % 3, d, i, j;
    if (km == 0)      { d = lp >> 12; int r = lp & 4095; i = r >> 6; j = r & 63; }
    else if (km == 1) { i = lp / 192; int r = lp % 192;  d = r >> 6; j = r & 63; }
    else              { i = lp / 192; int r = lp % 192;  j = r / 3;  d = r % 3;  }
    return d * HW + i * Ww + j;
}

// x_dbl[b,k,o,l] = sum_c xproj_w[k,o,c] * xs[b,k,c,l]
__global__ void k_xdbl(const float* __restrict__ stacked,
                       const float* __restrict__ xproj,
                       float* __restrict__ xdbl)
{
    int idx = blockIdx.x * blockDim.x + threadIdx.x;
    if (idx >= Bn * 6 * LTOT) return;
    int l = idx % LTOT;
    int k = (idx / LTOT) % 6;
    int b = idx / (6 * LTOT);
    int soff = scan_pos(k, l);
    const float* sb = stacked + (size_t)b * CDIM * 3 * HW + soff;
    float o[8] = {0, 0, 0, 0, 0, 0, 0, 0};
    for (int c = 0; c < CDIM; ++c) {
        float xv = sb[(size_t)c * 3 * HW];
        const float* wp = xproj + (size_t)k * 8 * CDIM + c;
#pragma unroll
        for (int oo = 0; oo < 8; ++oo) o[oo] += wp[oo * CDIM] * xv;
    }
    float* xo = xdbl + ((size_t)(b * 6 + k) * 8) * LTOT + l;
#pragma unroll
    for (int oo = 0; oo < 8; ++oo) xo[(size_t)oo * LTOT] = o[oo];
}

__global__ void k_zero(float* __restrict__ p, int n)
{
    int i = blockIdx.x * blockDim.x + threadIdx.x;
    if (i < n) p[i] = 0.f;
}

// Selective scan (n=1) per (b,k,c) channel; cross-merge via atomic accumulate
__global__ void k_scan(const float* __restrict__ stacked,
                       const float* __restrict__ xdbl,
                       const float* __restrict__ dt_w,   // (6,96,6)
                       const float* __restrict__ dt_b,   // (576,)
                       const float* __restrict__ A_logs, // (576,1)
                       const float* __restrict__ Ds,     // (576,)
                       float* __restrict__ msef)         // (B,96,3,H,W)
{
    int idx = blockIdx.x * blockDim.x + threadIdx.x;
    if (idx >= Bn * 6 * CDIM) return;
    int c = idx % CDIM;
    int k = (idx / CDIM) % 6;
    int b = idx / (6 * CDIM);
    int kd = k * CDIM + c;
    float A   = -expf(A_logs[kd]);
    float Dv  = Ds[kd];
    float dtb = dt_b[kd];
    const float* dtw = dt_w + (size_t)kd * 6;
    const float* xd  = xdbl + ((size_t)(b * 6 + k) * 8) * LTOT;
    const float* sb  = stacked + ((size_t)(b * CDIM + c) * 3) * HW;
    float* mb        = msef    + ((size_t)(b * CDIM + c) * 3) * HW;
    float h = 0.f;
    for (int l = 0; l < LTOT; ++l) {
        int soff = scan_pos(k, l);
        float u  = sb[soff];
        float dt = dtb;
#pragma unroll
        for (int r = 0; r < 6; ++r) dt += dtw[r] * xd[(size_t)r * LTOT + l];
        float Bs = xd[(size_t)6 * LTOT + l];
        float Cs = xd[(size_t)7 * LTOT + l];
        float delta = (dt > 20.f) ? dt : log1pf(expf(dt));
        h = expf(delta * A) * h + delta * Bs * u;
        atomicAdd(&mb[soff], h * Cs + u * Dv);
    }
}

// theta_shift (rotary): y = (x*cos + rot2(x)*sin) * scale
__global__ void k_theta(const float* __restrict__ x, long bstr_in,
                        const float* __restrict__ sinp, const float* __restrict__ cosp,
                        float* __restrict__ y, long bstr_out, float scale)
{
    int idx = blockIdx.x * blockDim.x + threadIdx.x;
    if (idx >= Bn * CDIM * HW) return;
    int p  = idx & (HW - 1);
    int ch = (idx >> 12) % CDIM;
    int b  = idx / (CDIM * HW);
    int cc = ch % 12;
    float sn = sinp[(size_t)p * 12 + cc];
    float cs = cosp[(size_t)p * 12 + cc];
    const float* xb = x + (size_t)b * bstr_in;
    float xv  = xb[(size_t)ch * HW + p];
    float rot = (cc & 1) ? xb[(size_t)(ch - 1) * HW + p]
                         : -xb[(size_t)(ch + 1) * HW + p];
    y[(size_t)b * bstr_out + (size_t)ch * HW + p] = (xv * cs + rot * sn) * scale;
}

// 3x3 neighborhood attention, 8 heads x 12 ch; v lives at kv channels 96..191
__global__ void k_na2d(const float* __restrict__ qt, const float* __restrict__ kt,
                       const float* __restrict__ kv, float* __restrict__ y)
{
    int idx = blockIdx.x * blockDim.x + threadIdx.x;
    if (idx >= Bn * 8 * HW) return;
    int j = idx & 63, i = (idx >> 6) & 63;
    int g = (idx >> 12) & 7;
    int b = idx >> 15;
    int ri = min(max(i - 1, 0), Hh - 3);
    int ci = min(max(j - 1, 0), Ww - 3);
    const size_t hb = ((size_t)b * CDIM + g * 12) * HW;
    float qv[12];
#pragma unroll
    for (int c = 0; c < 12; ++c) qv[c] = qt[hb + (size_t)c * HW + i * Ww + j];
    float e[9], mx = -1e30f;
    for (int u = 0; u < 3; ++u)
        for (int v = 0; v < 3; ++v) {
            size_t kp = hb + (size_t)(ri + u) * Ww + (ci + v);
            float d = 0.f;
#pragma unroll
            for (int c = 0; c < 12; ++c) d += qv[c] * kt[kp + (size_t)c * HW];
            e[u * 3 + v] = d;
            mx = fmaxf(mx, d);
        }
    float se = 0.f;
    for (int t = 0; t < 9; ++t) { e[t] = expf(e[t] - mx); se += e[t]; }
    float inv = 1.f / se;
    float out[12] = {0, 0, 0, 0, 0, 0, 0, 0, 0, 0, 0, 0};
    const size_t vb = ((size_t)b * 192 + 96 + g * 12) * HW;
    for (int u = 0; u < 3; ++u)
        for (int v = 0; v < 3; ++v) {
            float wgt = e[u * 3 + v] * inv;
            size_t vp = vb + (size_t)(ri + u) * Ww + (ci + v);
#pragma unroll
            for (int c = 0; c < 12; ++c) out[c] += wgt * kv[vp + (size_t)c * HW];
        }
#pragma unroll
    for (int c = 0; c < 12; ++c) y[hb + (size_t)c * HW + i * Ww + j] = out[c];
}

// ---------------------------------------------------------------------------
static inline int cdiv(int a, int b) { return (a + b - 1) / b; }

static void conv1x1(const float* X, const float* W, const float* bias,
                    const float* Res, float* Y, int Cout, hipStream_t s)
{
    int waves  = Bn * (Cout >> 5) * (HW >> 6);
    int blocks = cdiv(waves, GEMM_WAVES);
    k_conv1x1_wmma<<<blocks, 128, 0, s>>>(X, W, bias, Res, Y, Cout);
}

extern "C" void kernel_launch(void* const* d_in, const int* in_sizes, int n_in,
                              void* d_out, int out_size, void* d_ws, size_t ws_size,
                              hipStream_t stream)
{
    const float* art = (const float*)d_in[0];
    const float* pv  = (const float*)d_in[1];
    const float* dl  = (const float*)d_in[2];
    const float* gl  = (const float*)d_in[3];
    const float* sn  = (const float*)d_in[4];
    const float* cs  = (const float*)d_in[5];
    const float* dwc_dw_w = (const float*)d_in[6];
    const float* dwc_dw_b = (const float*)d_in[7];
    const float* dwc_gn_g = (const float*)d_in[8];
    const float* dwc_gn_b = (const float*)d_in[9];
    const float* dwc_pw_w = (const float*)d_in[10];
    const float* dwc_pw_b = (const float*)d_in[11];
    const float* dwc_ln_g = (const float*)d_in[12];
    const float* dwc_ln_b = (const float*)d_in[13];
    const float* norm_g   = (const float*)d_in[14];
    const float* norm_b   = (const float*)d_in[15];
    const float* op_w     = (const float*)d_in[16];
    const float* op_b     = (const float*)d_in[17];
    const float* le_c1_w  = (const float*)d_in[18];
    const float* le_c1_b  = (const float*)d_in[19];
    const float* le_gn1_g = (const float*)d_in[20];
    const float* le_gn1_b = (const float*)d_in[21];
    const float* le_dw_w  = (const float*)d_in[22];
    const float* le_dw_b  = (const float*)d_in[23];
    const float* le_gn2_g = (const float*)d_in[24];
    const float* le_gn2_b = (const float*)d_in[25];
    const float* le_c2_w  = (const float*)d_in[26];
    const float* le_c2_b  = (const float*)d_in[27];
    const float* le_gn3_g = (const float*)d_in[28];
    const float* le_gn3_b = (const float*)d_in[29];
    const float* xproj_w  = (const float*)d_in[30];
    const float* dt_w     = (const float*)d_in[31];
    const float* dt_b     = (const float*)d_in[32];
    const float* A_logs   = (const float*)d_in[33];
    const float* Ds       = (const float*)d_in[34];
    const float* q_w      = (const float*)d_in[35];
    const float* q_b      = (const float*)d_in[36];
    const float* kv_w     = (const float*)d_in[37];
    const float* kv_b     = (const float*)d_in[38];
    const float* lepe_w   = (const float*)d_in[39];
    const float* lepe_b   = (const float*)d_in[40];
    const float* op2_w    = (const float*)d_in[41];
    const float* op2_b    = (const float*)d_in[42];

    float* ws = (float*)d_ws;
    float* loc     = ws;                      // FEAT
    float* stacked = ws + (size_t)1 * FEAT;   // 3*FEAT, (B,96,3,H,W)
    float* msef    = ws + (size_t)4 * FEAT;   // 3*FEAT
    float* xdbl    = ws + (size_t)7 * FEAT;   // 1.5*FEAT (B*6*8*L)
    float* t0      = ws + (size_t)8 * FEAT + FEAT / 2;
    float* t1      = t0 + FEAT;
    float* t2      = t1 + FEAT;
    float* kvb     = t2 + FEAT;               // 2*FEAT (B,192,H,W)
    float* kt      = kvb + 2 * (size_t)FEAT;  // FEAT
    float* lepe    = kt + FEAT;               // FEAT
    float* qt      = lepe + FEAT;             // FEAT
    float* br      = qt + FEAT;               // 3*FEAT (art2,pv2,dl2)
    float* out     = (float*)d_out;

    const long FB = (long)CDIM * HW;
    const int EW = cdiv(Bn * CDIM * HW, 256);

    // ---- local_en_block(gl) -> loc ----
    conv1x1(gl, le_c1_w, le_c1_b, nullptr, t0, CDIM, stream);
    k_gn<<<Bn * CDIM, 256, 0, stream>>>(t0, le_gn1_g, le_gn1_b, t1, 0);
    k_dwconv<<<EW, 256, 0, stream>>>(t1, le_dw_w, le_dw_b, t2, 5, 2, FB, FB);
    k_gn<<<Bn * CDIM, 256, 0, stream>>>(t2, le_gn2_g, le_gn2_b, t0, 1);   // + GELU
    conv1x1(t0, le_c2_w, le_c2_b, nullptr, t1, CDIM, stream);
    k_gn<<<Bn * CDIM, 256, 0, stream>>>(t1, le_gn3_g, le_gn3_b, loc, 0);

    // ---- dwc_block(art/pv/dl) -> silu -> stacked (B,96,3,H,W) ----
    const float* mods[3] = {art, pv, dl};
    for (int m = 0; m < 3; ++m) {
        k_dwconv<<<EW, 256, 0, stream>>>(mods[m], dwc_dw_w, dwc_dw_b, t0, 3, 1, FB, FB);
        k_gn<<<Bn * CDIM, 256, 0, stream>>>(t0, dwc_gn_g, dwc_gn_b, t1, 0);
        conv1x1(t1, dwc_pw_w, dwc_pw_b, nullptr, t2, CDIM, stream);
        k_ln2d<<<cdiv(Bn * HW, 256), 256, 0, stream>>>(
            t2, FB, HW, nullptr, dwc_ln_g, dwc_ln_b,
            stacked + (size_t)m * HW, 3 * FB, 3 * (long)HW, /*silu=*/1);
    }

    // ---- ss2d: zero msef, x_dbl, selective scan + cross-merge ----
    k_zero<<<cdiv(3 * FEAT, 256), 256, 0, stream>>>(msef, 3 * FEAT);
    k_xdbl<<<cdiv(Bn * 6 * LTOT, 256), 256, 0, stream>>>(stacked, xproj_w, xdbl);
    k_scan<<<cdiv(Bn * 6 * CDIM, 256), 256, 0, stream>>>(
        stacked, xdbl, dt_w, dt_b, A_logs, Ds, msef);

    // ---- branch: x + conv1x1(loc + ln2d(msef[:,:,m])) ----
    for (int m = 0; m < 3; ++m) {
        k_ln2d<<<cdiv(Bn * HW, 256), 256, 0, stream>>>(
            msef + (size_t)m * HW, 3 * FB, 3 * (long)HW, loc, norm_g, norm_b,
            t0, FB, HW, 0);
        conv1x1(t0, op_w, op_b, /*Res=*/mods[m], br + (size_t)m * FEAT,
                CDIM, stream);
    }

    // ---- gl_fusion shared: kv = conv(gl), kt = theta(k), lepe = dw5x5(v) ----
    conv1x1(gl, kv_w, kv_b, nullptr, kvb, 192, stream);
    k_theta<<<EW, 256, 0, stream>>>(kvb, 192L * HW, sn, cs, kt, FB, 1.0f);
    k_dwconv<<<EW, 256, 0, stream>>>(kvb + (size_t)CDIM * HW, lepe_w, lepe_b,
                                     lepe, 5, 2, 192L * HW, FB);

    // ---- per-branch fusion -> d_out ----
    for (int m = 0; m < 3; ++m) {
        const float* xq = br + (size_t)m * FEAT;
        conv1x1(xq, q_w, q_b, nullptr, t0, CDIM, stream);
        k_theta<<<EW, 256, 0, stream>>>(t0, FB, sn, cs, qt, FB, SCALE_Q);
        k_na2d<<<cdiv(Bn * 8 * HW, 256), 256, 0, stream>>>(qt, kt, kvb, t1);
        k_add<<<cdiv(FEAT, 256), 256, 0, stream>>>(t1, lepe, t2, FEAT);
        conv1x1(t2, op2_w, op2_b, nullptr, out + (size_t)m * FEAT, CDIM, stream);
    }
}
